// Conv2d_42417097016539
// MI455X (gfx1250) — matrix operand order repaired and checked by reading, not compiled, not run on hardware
//
#include <hip/hip_runtime.h>

// Implicit-GEMM 3x3 conv (NCHW, pad=1, stride=1) for MI455X / gfx1250.
//   M = 256 outch, Npix = 32*56*56 = 100352, Kred = 128*9 = 1152
// fp32 in/out; f16 WMMA with 2-term f16 split: acc += Ah*Bh + Ah*Bl + Al*Bh.
//
// Tap-major reduction r' = (kh*3+kw)*128 + c: tap decode + pad predicate are
// per-chunk. Both operands are pre-converted once into f16 hi/lo planes
// (weights tap-major [m][tap*128+c]; x c-innermost [n*HW+hw][c]) so the main
// loop stages tiles as pure b128 copies with zero conversion VALU.
// Block tile 128x128, 512 threads / 16 waves, wave tile 32x32 (4 accums),
// single 40 KB LDS buffer. Staging registers are reused across iterations
// (stores consume VGPRs at issue -> no double set, no copies, no spills);
// main loop is #pragma unroll 1 to keep live ranges within one iteration.
// Workspace tiers: MODE 2 (x+w packed) / 1 (w packed) / 0 (raw fallback).

typedef __attribute__((ext_vector_type(16))) _Float16 v16h;
typedef __attribute__((ext_vector_type(8)))  float    v8f;

#define IMG_W   56
#define IMG_HW  3136
#define N_IMG   32
#define C_IN    128
#define K_OUT   256
#define K_RED   1152
#define TILE_M  128
#define TILE_N  128
#define TILE_K  32
#define NITER   (K_RED / TILE_K)             // 36 = 9 taps x 4 c-chunks
#define LDS_STR 40                            // halfs/row = 80B, conflict-free
#define A_PLANE (TILE_M * LDS_STR)            // 5120 halfs
#define B_PLANE (TILE_N * LDS_STR)            // 5120 halfs
#define AH_OFF  0
#define AL_OFF  (A_PLANE)
#define BH_OFF  (2 * A_PLANE)
#define BL_OFF  (2 * A_PLANE + B_PLANE)
#define BUF_HALFS (2 * A_PLANE + 2 * B_PLANE) // 20480 halfs = 40960 B
#define WP_HALFS ((size_t)K_OUT * K_RED)      // 294912 halfs per plane
#define XP_HALFS ((size_t)N_IMG * C_IN * IMG_HW) // 12845056 halfs per plane

union Frag  { v16h h; uint4 q[2]; };
union Pack8 { _Float16 h[8]; uint4 q; };
union ARegs { uint4 q[2]; float f[8]; };
union BRegs { uint4 q[2]; float f[8]; };

// A fragment: 16-bit A 16x32 layout (ISA 7.12.2):
//   lane L<16 : M=L,    K={0..7} v0..v3, {16..23} v4..v7
//   lane L>=16: M=L-16, K={8..15} and {24..31}
__device__ inline v16h load_fragA(const _Float16* tile, int lane) {
  const int row = lane & 15;
  const int k0  = (lane & 16) ? 8 : 0;
  const uint4* p = (const uint4*)(tile + row * LDS_STR);
  Frag f;
  f.q[0] = p[k0 >> 3];
  f.q[1] = p[(16 + k0) >> 3];
  return f.h;
}

// B fragment: 16-bit B 32x16 layout: lane L<16: N=L, K={0..7},{16..23}; L>=16: K={8..15},{24..31}.
__device__ inline v16h load_fragB(const _Float16* tile, int lane) {
  const int row = lane & 15;
  const int q0  = (lane & 16) ? 1 : 0;
  const uint4* p = (const uint4*)(tile + row * LDS_STR);
  Frag f;
  f.q[0] = p[q0];
  f.q[1] = p[q0 + 2];
  return f.h;
}

// Pre-pack weights OIHW -> tap-major f16 hi/lo: wp[m][tap*128 + c].
__global__ __launch_bounds__(256)
void prepack_w(const float* __restrict__ w, _Float16* __restrict__ wph,
               _Float16* __restrict__ wpl) {
  const unsigned idx = blockIdx.x * 256u + threadIdx.x;   // 0..294911
  const unsigned m   = idx / K_RED;
  const unsigned r   = idx - m * K_RED;
  const unsigned tap = r >> 7;
  const unsigned c   = r & 127u;
  const float v = w[(size_t)m * K_RED + c * 9u + tap];
  const _Float16 h = (_Float16)v;
  wph[idx] = h;
  wpl[idx] = (_Float16)(v - (float)h);
}

// Pre-pack x NCHW -> c-innermost f16 hi/lo: xp[n*HW + hw][c].
__global__ __launch_bounds__(256)
void prepack_x(const float* __restrict__ x, _Float16* __restrict__ xph,
               _Float16* __restrict__ xpl) {
  const unsigned pix = blockIdx.x * 256u + threadIdx.x;   // 0..100351
  const unsigned c0  = blockIdx.y * 8u;
  const unsigned n   = pix / IMG_HW;
  const unsigned hw  = pix - n * IMG_HW;
  const float* src = x + ((size_t)n * C_IN + c0) * IMG_HW + hw;  // coalesced
  Pack8 hp, lp;
#pragma unroll
  for (int j = 0; j < 8; ++j) {
    const float v = src[(size_t)j * IMG_HW];
    const _Float16 h = (_Float16)v;
    hp.h[j] = h;
    lp.h[j] = (_Float16)(v - (float)h);
  }
  const size_t dst = (size_t)pix * C_IN + c0;
  *(uint4*)(xph + dst) = hp.q;
  *(uint4*)(xpl + dst) = lp.q;
}

template <int MODE>   // 0 = raw, 1 = w packed, 2 = w + x packed
__global__ __launch_bounds__(512)
void conv3x3_wmma_f16split(const float* __restrict__ x,
                           const float* __restrict__ w,
                           const _Float16* __restrict__ wph,
                           const _Float16* __restrict__ wpl,
                           const _Float16* __restrict__ xph,
                           const _Float16* __restrict__ xpl,
                           float* __restrict__ out) {
  __shared__ __attribute__((aligned(16))) _Float16 smem[BUF_HALFS]; // 40 KB

  const int t      = threadIdx.x;
  const int lane   = t & 31;
  const int wid    = t >> 5;      // 0..15
  const int wave_m = wid >> 2;    // 0..3 -> 32-row slice of M tile
  const int wave_n = wid & 3;     // 0..3 -> 32-col slice of N tile

  const int      m_block  = blockIdx.y * TILE_M;
  const unsigned pix_base = blockIdx.x * TILE_N;

  // ---- A staging: 128 rows x 32 k, 8 halfs/thread per plane (1 x b128)
  const int aRow = t >> 2;            // 0..127
  const int aK0  = (t & 3) * 8;       // 0,8,16,24

  // ---- B staging: 128 rows x 32 k, 8 elems/thread; lanes contiguous in pixel
  const int bRow = t & 127;           // pixel in tile
  const int bK0  = (t >> 7) * 8;      // 0,8,16,24 (channel offset in chunk)
  const unsigned p    = pix_base + (unsigned)bRow;
  const unsigned nimg = p / IMG_HW;          // tile may straddle images
  const unsigned hw   = p - nimg * IMG_HW;
  const int oh = (int)(hw / IMG_W);
  const int ow = (int)(hw - (unsigned)oh * IMG_W);
  const float* xImg = x + (size_t)nimg * C_IN * IMG_HW;

  auto loadA = [&](int i, ARegs& r) {
    if (MODE >= 1) {
      const size_t off = (size_t)(m_block + aRow) * K_RED + i * TILE_K + aK0;
      r.q[0] = *(const uint4*)(wph + off);
      r.q[1] = *(const uint4*)(wpl + off);
    } else {
      const int tap = i >> 2;
      const int c0  = (i & 3) * TILE_K + aK0;
      const float* wp = w + (size_t)(m_block + aRow) * K_RED + c0 * 9 + tap;
#pragma unroll
      for (int j = 0; j < 8; ++j) r.f[j] = wp[(size_t)j * 9];
    }
  };

  auto storeA = [&](ARegs& r) {
    uint4* dh = (uint4*)(smem + AH_OFF + aRow * LDS_STR + aK0);
    uint4* dl = (uint4*)(smem + AL_OFF + aRow * LDS_STR + aK0);
    if (MODE >= 1) {
      dh[0] = r.q[0];
      dl[0] = r.q[1];
    } else {
      Pack8 hp, lp;
#pragma unroll
      for (int j = 0; j < 8; ++j) {
        const float va = r.f[j];
        const _Float16 ha = (_Float16)va;
        hp.h[j] = ha; lp.h[j] = (_Float16)(va - (float)ha);
      }
      dh[0] = hp.q;
      dl[0] = lp.q;
    }
  };

  auto loadB = [&](int i, BRegs& r) {
    const int tap = i >> 2;                 // uniform per chunk
    const int kh  = tap / 3;
    const int kw  = tap - kh * 3;
    const int c0  = (i & 3) * TILE_K + bK0;
    const int ih  = oh - 1 + kh;
    const int iw  = ow - 1 + kw;
    const bool valid = ((unsigned)ih < 56u) && ((unsigned)iw < 56u);
    if (MODE == 2) {
      const uint4 z = {0u, 0u, 0u, 0u};
      r.q[0] = z; r.q[1] = z;
      if (valid) {   // one predicate per chunk, pure b128 copies
        const size_t base =
            ((size_t)nimg * IMG_HW + (unsigned)(ih * IMG_W + iw)) * C_IN + c0;
        r.q[0] = *(const uint4*)(xph + base);
        r.q[1] = *(const uint4*)(xpl + base);
      }
    } else {
#pragma unroll
      for (int j = 0; j < 8; ++j) r.f[j] = 0.f;
      if (valid) {
        const float* src = xImg + (size_t)c0 * IMG_HW + ih * IMG_W + iw;
#pragma unroll
        for (int j = 0; j < 8; ++j) r.f[j] = src[(size_t)j * IMG_HW];
      }
    }
  };

  auto storeB = [&](BRegs& r) {
    uint4* dh = (uint4*)(smem + BH_OFF + bRow * LDS_STR + bK0);
    uint4* dl = (uint4*)(smem + BL_OFF + bRow * LDS_STR + bK0);
    if (MODE == 2) {
      dh[0] = r.q[0];
      dl[0] = r.q[1];
    } else {
      Pack8 hp, lp;
#pragma unroll
      for (int j = 0; j < 8; ++j) {
        const _Float16 h = (_Float16)r.f[j];
        hp.h[j] = h;
        lp.h[j] = (_Float16)(r.f[j] - (float)h);
      }
      dh[0] = hp.q;
      dl[0] = lp.q;
    }
  };

  v8f acc[2][2] = {{{}, {}}, {{}, {}}};

  // Single staging register set, reused every iteration: DS/VMEM stores read
  // their data VGPRs at issue, so overwriting them with the next chunk's
  // global loads right after the barrier is safe (no copies, short liveness).
  ARegs ar;  BRegs br;
  loadA(0, ar);
  loadB(0, br);

#pragma unroll 1
  for (int i = 0; i < NITER; ++i) {
    storeA(ar);
    storeB(br);
    __syncthreads();   // publish chunk i

    if (i + 1 < NITER) { loadA(i + 1, ar); loadB(i + 1, br); }  // overlap WMMA

    const _Float16* aTh = smem + AH_OFF + wave_m * 32 * LDS_STR;
    const _Float16* aTl = smem + AL_OFF + wave_m * 32 * LDS_STR;
    const _Float16* bTh = smem + BH_OFF + wave_n * 32 * LDS_STR;
    const _Float16* bTl = smem + BL_OFF + wave_n * 32 * LDS_STR;
    v16h aHi[2], aLo[2], bHi[2], bLo[2];
#pragma unroll
    for (int s = 0; s < 2; ++s) {
      aHi[s] = load_fragA(aTh + s * 16 * LDS_STR, lane);
      aLo[s] = load_fragA(aTl + s * 16 * LDS_STR, lane);
      bHi[s] = load_fragB(bTh + s * 16 * LDS_STR, lane);
      bLo[s] = load_fragB(bTl + s * 16 * LDS_STR, lane);
    }

    // term-outer order: consecutive WMMAs hit different accumulators.
#pragma unroll
    for (int mi = 0; mi < 2; ++mi)
#pragma unroll
      for (int ni = 0; ni < 2; ++ni)
        acc[mi][ni] = __builtin_amdgcn_wmma_f32_16x16x32_f16(
            false, aHi[mi], false, bHi[ni], (short)0, acc[mi][ni], false, false);
#pragma unroll
    for (int mi = 0; mi < 2; ++mi)
#pragma unroll
      for (int ni = 0; ni < 2; ++ni)
        acc[mi][ni] = __builtin_amdgcn_wmma_f32_16x16x32_f16(
            false, aHi[mi], false, bLo[ni], (short)0, acc[mi][ni], false, false);
#pragma unroll
    for (int mi = 0; mi < 2; ++mi)
#pragma unroll
      for (int ni = 0; ni < 2; ++ni)
        acc[mi][ni] = __builtin_amdgcn_wmma_f32_16x16x32_f16(
            false, aLo[mi], false, bHi[ni], (short)0, acc[mi][ni], false, false);

    __syncthreads();   // reads of chunk i done before next iter's stores
  }

  // ---- epilogue: 16x16 fp32 C/D layout: lane<16: N=lane, M=r; else M=r+8.
  const int nloc  = lane & 15;
  const int mhalf = (lane >> 4) << 3;
#pragma unroll
  for (int ni = 0; ni < 2; ++ni) {
    const unsigned pp  = pix_base + (unsigned)(wave_n * 32 + ni * 16 + nloc);
    const unsigned nim = pp / IMG_HW;
    const unsigned hwp = pp - nim * IMG_HW;
    float* oBase = out + (size_t)nim * K_OUT * IMG_HW + hwp;
#pragma unroll
    for (int mi = 0; mi < 2; ++mi) {
      const int mbase = m_block + wave_m * 32 + mi * 16 + mhalf;
      float* o = oBase + (size_t)mbase * IMG_HW;
#pragma unroll
      for (int r = 0; r < 8; ++r) o[(size_t)r * IMG_HW] = acc[mi][ni][r];
    }
  }
}

extern "C" void kernel_launch(void* const* d_in, const int* in_sizes, int n_in,
                              void* d_out, int out_size, void* d_ws, size_t ws_size,
                              hipStream_t stream) {
  const float* x = (const float*)d_in[0];   // [32,128,56,56]
  const float* w = (const float*)d_in[1];   // [256,128,3,3]
  float*       o = (float*)d_out;           // [32,256,56,56]

  dim3 grid(100352 / TILE_N, K_OUT / TILE_M);   // (784, 2)
  const size_t needW = WP_HALFS * 2 * sizeof(_Float16);             // ~1.18 MB
  const size_t needX = needW + XP_HALFS * 2 * sizeof(_Float16);     // ~52.6 MB

  if (ws_size >= needX) {
    _Float16* wph = (_Float16*)d_ws;
    _Float16* wpl = wph + WP_HALFS;
    _Float16* xph = wpl + WP_HALFS;
    _Float16* xpl = xph + XP_HALFS;
    prepack_w<<<(unsigned)(WP_HALFS / 256), 256, 0, stream>>>(w, wph, wpl);
    prepack_x<<<dim3(100352 / 256, C_IN / 8), 256, 0, stream>>>(x, xph, xpl);
    conv3x3_wmma_f16split<2><<<grid, 512, 0, stream>>>(x, w, wph, wpl, xph, xpl, o);
  } else if (ws_size >= needW) {
    _Float16* wph = (_Float16*)d_ws;
    _Float16* wpl = wph + WP_HALFS;
    prepack_w<<<(unsigned)(WP_HALFS / 256), 256, 0, stream>>>(w, wph, wpl);
    conv3x3_wmma_f16split<1><<<grid, 512, 0, stream>>>(x, w, wph, wpl, nullptr, nullptr, o);
  } else {
    conv3x3_wmma_f16split<0><<<grid, 512, 0, stream>>>(x, w, nullptr, nullptr, nullptr, nullptr, o);
  }
  (void)in_sizes; (void)n_in; (void)out_size;
}
